// FeedForward_20607253086906
// MI455X (gfx1250) — matrix operand order repaired and checked by reading, not compiled, not run on hardware
//
#include <hip/hip_runtime.h>
#include <hip/hip_bf16.h>

typedef __attribute__((ext_vector_type(16))) __bf16 v16bf;
typedef __attribute__((ext_vector_type(8)))  float  v8f;
typedef unsigned int uint32x4 __attribute__((ext_vector_type(4)));
typedef int          int32x8  __attribute__((ext_vector_type(8)));
typedef int          int32x4  __attribute__((ext_vector_type(4)));

#define ZERO8 ((v8f){0.f,0.f,0.f,0.f,0.f,0.f,0.f,0.f})

// ---------------------------------------------------------------------------
// Geometry:
//   IRR_IN  = [(256,0),(128,1),(64,2)]   dim 960
//   IRR_H   = [(512,0),(256,1),(128,2)]  dim 1920
//   N = 32768 rows, out dim 960.
// Packed bf16 weight arena in d_ws (path-norm scale folded in), fragment-
// native layout:  frag(kt, ct) = 32 lanes x 16 contiguous bf16, where
// element(lane, e) = W[kt*32 + (lane>>4)*16 + e][ct*16 + (lane&15)] * scale.
// Arena offsets (bf16 elems):
//   s1: 0 / 131072 / 163840        (172032)
//   s2: 172032 / 434176 / 499712   (344064)
//   s3: 516096 / 647168 / 679936   (172032)
//   sb: 688128 / 753664 / 770048   ( 86016)   total 774144 -> 1548288 bytes
// stats (704 f32) live after the arena.
// ---------------------------------------------------------------------------

__device__ __forceinline__ __bf16 f2bf(float f) {
    union { float f; unsigned u; } uf; uf.f = f;
    union { unsigned short s; __bf16 b; } ub;
    ub.s = (unsigned short)(uf.u >> 16);
    return ub.b;
}

// ---------------------------------------------------------------------------
// Weight repack: f32 [K][MO] -> bf16 WMMA-B-fragment order, scale folded in.
// ---------------------------------------------------------------------------
__global__ void pack_weights(const float* __restrict__ w1, const float* __restrict__ w2,
                             const float* __restrict__ w3, const float* __restrict__ wb,
                             __bf16* __restrict__ wp)
{
    const int   KK [12] = {256,128,64, 512,256,128, 512,256,128, 256,128,64};
    const int   MO [12] = {512,256,128, 512,256,128, 256,128,64, 256,128,64};
    const int   SRC[12] = {0,131072,163840, 0,262144,327680, 0,131072,163840, 0,65536,81920};
    const int   DST[12] = {0,131072,163840, 172032,434176,499712, 516096,647168,679936,
                           688128,753664,770048};
    const float SC [12] = {0.0625f,0.08838834764f,0.125f,
                           0.04419417382f,0.0625f,0.08838834764f,
                           0.04419417382f,0.0625f,0.08838834764f,
                           0.0625f,0.08838834764f,0.125f};
    const int e12 = blockIdx.y;
    const int K = KK[e12], mo = MO[e12];
    const int sz = K * mo;
    const int idx = blockIdx.x * blockDim.x + threadIdx.x;
    if (idx >= sz) return;

    const float* src;
    switch (e12 / 3) { case 0: src = w1; break; case 1: src = w2; break;
                       case 2: src = w3; break; default: src = wb; break; }

    const int e    = idx & 15;
    const int lane = (idx >> 4) & 31;
    const int fct  = idx >> 9;            // fragment index = kt*nct + ct
    const int nct  = mo >> 4;
    const int kt   = fct / nct;
    const int ct   = fct - kt * nct;
    const int k    = kt * 32 + (lane >> 4) * 16 + e;
    const int col  = ct * 16 + (lane & 15);
    wp[DST[e12] + idx] = f2bf(src[SRC[e12] + k * mo + col] * SC[e12]);
}

// ---------------------------------------------------------------------------
// A-fragment load (contiguous, matches packed B): lanes 0-15 hold K k0+0..15,
// lanes 16-31 hold k0+16..31, row = lane&15.
// ---------------------------------------------------------------------------
__device__ __forceinline__ v16bf load_afrag(const __bf16* __restrict__ ap, int kh)
{
    v16bf a;
#pragma unroll
    for (int e = 0; e < 16; ++e) {
        a[e] = ap[kh * 16 + e];
    }
    return a;
}

// NC irrep components share each B fragment.
template<int NC>
__device__ __forceinline__ void gemm_block(v8f (&acc)[NC],
                                           const __bf16* __restrict__ Abase, int K,
                                           const __bf16* __restrict__ Wp, int nct,
                                           int ct, int lane)
{
    const int row = lane & 15;
    const int kh  = lane >> 4;
    for (int k0 = 0, kt = 0; k0 < K; k0 += 32, ++kt) {
        const v16bf b = *(const v16bf*)(Wp + ((size_t)(kt * nct + ct) * 32 + lane) * 16);
#pragma unroll
        for (int c = 0; c < NC; ++c) {
            v16bf a = load_afrag(Abase + c * (16 * K) + row * K + k0, kh);
            acc[c] = __builtin_amdgcn_wmma_f32_16x16x32_bf16(
                false, a, false, b, (short)0, acc[c], false, false);
        }
    }
}

// NC==1 double-width variant: one A fragment feeds two adjacent column tiles
// (B fragments are contiguous in the packed arena) -> halves LDS A traffic
// on the l=0 GEMMs that carry most of the FLOPs.
__device__ __forceinline__ void gemm_block_x2(v8f (&acc)[2],
                                              const __bf16* __restrict__ Abase, int K,
                                              const __bf16* __restrict__ Wp, int nct,
                                              int ct0, int lane)
{
    const int row = lane & 15;
    const int kh  = lane >> 4;
    for (int k0 = 0, kt = 0; k0 < K; k0 += 32, ++kt) {
        const __bf16* fb = Wp + ((size_t)(kt * nct + ct0) * 32 + lane) * 16;
        const v16bf b0 = *(const v16bf*)fb;
        const v16bf b1 = *(const v16bf*)(fb + 512);      // next ct fragment
        v16bf a = load_afrag(Abase + row * K + k0, kh);
        acc[0] = __builtin_amdgcn_wmma_f32_16x16x32_bf16(
            false, a, false, b0, (short)0, acc[0], false, false);
        acc[1] = __builtin_amdgcn_wmma_f32_16x16x32_bf16(
            false, a, false, b1, (short)0, acc[1], false, false);
    }
}

// NormActivation f * sigmoid(|f|)/|f| across NC components, bf16-store to LDS.
template<int NC>
__device__ __forceinline__ void norm_store(v8f (&acc)[NC],
                                           __bf16* __restrict__ Hbase, int mo,
                                           int colbase, int lane)
{
    const int col = colbase + (lane & 15);
    const int rh  = (lane >> 4) * 8;
#pragma unroll
    for (int j = 0; j < 8; ++j) {
        float n2 = 1e-8f;
#pragma unroll
        for (int c = 0; c < NC; ++c) n2 += acc[c][j] * acc[c][j];
        float n = sqrtf(n2);
        float s = 1.f / ((1.f + __expf(-n)) * n);
        const int M = rh + j;
#pragma unroll
        for (int c = 0; c < NC; ++c)
            Hbase[c * (16 * mo) + M * mo + col] = f2bf(acc[c][j] * s);
    }
}

// Scalar (l=0) norm-act + store.
__device__ __forceinline__ void norm_store1(v8f& acc, __bf16* __restrict__ Hbase,
                                            int mo, int colbase, int lane)
{
    const int col = colbase + (lane & 15);
    const int rh  = (lane >> 4) * 8;
#pragma unroll
    for (int j = 0; j < 8; ++j) {
        float v = acc[j];
        float n = sqrtf(v * v + 1e-8f);
        float s = 1.f / ((1.f + __expf(-n)) * n);
        Hbase[(rh + j) * mo + col] = f2bf(v * s);
    }
}

// l=0 epilogue: write pre-BN output + mean/var stats.
__device__ __forceinline__ void out_store1(v8f& acc, float* __restrict__ y, int row0,
                                           int colbase, float* __restrict__ stats, int lane)
{
    const int col = colbase + (lane & 15);
    const int rh  = (lane >> 4) * 8;
    float s1 = 0.f, s2 = 0.f;
#pragma unroll
    for (int j = 0; j < 8; ++j) {
        float v = acc[j];
        y[(size_t)(row0 + rh + j) * 960 + col] = v;
        s1 += v; s2 += v * v;
    }
    atomicAdd(&stats[col], s1);
    atomicAdd(&stats[256 + col], s2);
}

// l>0 epilogue: write pre-BN output + squared-norm stats.
template<int NC>
__device__ __forceinline__ void out_store(v8f (&acc)[NC],
                                          float* __restrict__ y, int row0,
                                          int flatoff, int colbase,
                                          float* __restrict__ stats, int statoff,
                                          int lane)
{
    const int col = colbase + (lane & 15);
    const int rh  = (lane >> 4) * 8;
    float s2 = 0.f;
#pragma unroll
    for (int j = 0; j < 8; ++j) {
        const int r = row0 + rh + j;
#pragma unroll
        for (int c = 0; c < NC; ++c) {
            float v = acc[c][j];
            y[(size_t)r * 960 + flatoff + col * NC + c] = v;
            s2 += v * v;
        }
    }
    atomicAdd(&stats[statoff + col], s2);
}

// ---------------------------------------------------------------------------
// Fused chain.  LDS byte map (153600 B dynamic):
//   [     0,  30720)  xs  bf16 x-tile in irrep layout
//   [ 30720,  92160)  h1  bf16
//   [ 92160, 153600)  h2  bf16  -- doubles as raw f32 x staging (TDM target)
// ---------------------------------------------------------------------------
__global__ void __launch_bounds__(256, 1)
ff_fused(const float* __restrict__ x, const __bf16* __restrict__ wp,
         float* __restrict__ y, float* __restrict__ stats)
{
    extern __shared__ __bf16 smem[];
    __bf16* xs   = smem;
    __bf16* h1   = smem + 15360;
    __bf16* h2   = smem + 15360 + 30720;
    float*  xraw = (float*)h2;                   // staging aliases h2

    const int tid  = threadIdx.x;
    const int lane = tid & 31;
    const int wave = __builtin_amdgcn_readfirstlane(tid >> 5);
    const int row0 = blockIdx.x * 16;

    // ---- stage 0a: TDM DMA of the raw 16x960 f32 tile into LDS staging.
#if __has_builtin(__builtin_amdgcn_tensor_load_to_lds)
    if (wave == 0) {
        unsigned long long ga = (unsigned long long)(const void*)(x + (size_t)row0 * 960);
        unsigned lds_off = (unsigned)(unsigned long long)(void*)xraw;  // addr[31:0] = LDS byte addr
        uint32x4 g0;
        g0[0] = 1u;                                    // count=1, user-mode D#
        g0[1] = lds_off;                               // lds_addr
        g0[2] = (unsigned)(ga & 0xffffffffu);          // global_addr[31:0]
        g0[3] = (unsigned)((ga >> 32) & 0x1ffffffu) | (2u << 30);  // addr[56:32] | type=2
        int32x8 g1;
        g1[0] = (int)(2u << 16);                       // data_size = 4 bytes
        g1[1] = (int)(960u << 16);                     // tensor_dim0[15:0] = 960
        g1[2] = (int)(32768u << 16);                   // dim0 hi=0, tensor_dim1[15:0]
        g1[3] = (int)(960u << 16);                     // dim1 hi=0, tile_dim0 = 960
        g1[4] = 16;                                    // tile_dim1 = 16 rows
        g1[5] = 960;                                   // tensor_dim0_stride = 960
        g1[6] = 0; g1[7] = 0;
        int32x4 gz = {0, 0, 0, 0};
#if defined(__clang_major__) && __clang_major__ >= 23
        int32x8 gz8 = {0, 0, 0, 0, 0, 0, 0, 0};
        __builtin_amdgcn_tensor_load_to_lds(g0, g1, gz, gz, gz8, 0);
#else
        __builtin_amdgcn_tensor_load_to_lds(g0, g1, gz, gz, 0);
#endif
        __builtin_amdgcn_s_wait_tensorcnt(0);
    }
#else
    for (int idx = tid; idx < 16 * 960; idx += 256)
        xraw[idx] = x[(size_t)row0 * 960 + idx];
#endif
    __syncthreads();

    // ---- stage 0b: convert + scatter into per-irrep [comp][row][chan] bf16.
    for (int idx = tid; idx < 16 * 960; idx += 256) {
        const int r   = idx / 960;
        const int col = idx - r * 960;
        const float v = xraw[idx];
        int base, mi, u, i;
        if (col < 256)      { base = 0;     mi = 256; u = col;        i = 0; }
        else if (col < 640) { int c2 = col - 256; base = 4096;  mi = 128; u = c2 / 3; i = c2 - 3 * u; }
        else                { int c2 = col - 640; base = 10240; mi = 64;  u = c2 / 5; i = c2 - 5 * u; }
        xs[base + i * 16 * mi + r * mi + u] = f2bf(v);
    }
    __syncthreads();

    // ---- stage 1: h1 = norm_act(linear(x, w1))   [960 -> 1920]
    // tasks: 16 double-tiles (l=0) + 16 (l=1) + 8 (l=2) = 40 -> 5 per wave
    for (int t = wave; t < 40; t += 8) {
        if (t < 16) {
            v8f acc[2] = {ZERO8, ZERO8};
            gemm_block_x2(acc, xs + 0,     256, wp + 0,      32, t * 2, lane);
            norm_store1(acc[0], h1 + 0, 512, t * 32,      lane);
            norm_store1(acc[1], h1 + 0, 512, t * 32 + 16, lane);
        } else if (t < 32) {
            const int ct = t - 16;
            v8f acc[3] = {ZERO8, ZERO8, ZERO8};
            gemm_block<3>(acc, xs + 4096,  128, wp + 131072, 16, ct, lane);
            norm_store<3>(acc, h1 + 8192,  256, ct * 16, lane);
        } else {
            const int ct = t - 32;
            v8f acc[5] = {ZERO8, ZERO8, ZERO8, ZERO8, ZERO8};
            gemm_block<5>(acc, xs + 10240, 64,  wp + 163840, 8,  ct, lane);
            norm_store<5>(acc, h1 + 20480, 128, ct * 16, lane);
        }
    }
    __syncthreads();

    // ---- stage 2: h2 = norm_act(linear(h1, w2))  [1920 -> 1920]
    for (int t = wave; t < 40; t += 8) {
        if (t < 16) {
            v8f acc[2] = {ZERO8, ZERO8};
            gemm_block_x2(acc, h1 + 0,     512, wp + 172032, 32, t * 2, lane);
            norm_store1(acc[0], h2 + 0, 512, t * 32,      lane);
            norm_store1(acc[1], h2 + 0, 512, t * 32 + 16, lane);
        } else if (t < 32) {
            const int ct = t - 16;
            v8f acc[3] = {ZERO8, ZERO8, ZERO8};
            gemm_block<3>(acc, h1 + 8192,  256, wp + 434176, 16, ct, lane);
            norm_store<3>(acc, h2 + 8192,  256, ct * 16, lane);
        } else {
            const int ct = t - 32;
            v8f acc[5] = {ZERO8, ZERO8, ZERO8, ZERO8, ZERO8};
            gemm_block<5>(acc, h1 + 20480, 128, wp + 499712, 8,  ct, lane);
            norm_store<5>(acc, h2 + 20480, 128, ct * 16, lane);
        }
    }
    __syncthreads();

    // ---- stage 3: y = linear(h2, w3) + linear(x, wb) ; stats accumulate
    // tasks: 8 double-tiles (l=0) + 8 (l=1) + 4 (l=2) = 20
    for (int t = wave; t < 20; t += 8) {
        if (t < 8) {
            v8f acc[2] = {ZERO8, ZERO8};
            gemm_block_x2(acc, h2 + 0,     512, wp + 516096, 16, t * 2, lane);
            gemm_block_x2(acc, xs + 0,     256, wp + 688128, 16, t * 2, lane);
            out_store1(acc[0], y, row0, t * 32,      stats, lane);
            out_store1(acc[1], y, row0, t * 32 + 16, stats, lane);
        } else if (t < 16) {
            const int ct = t - 8;
            v8f acc[3] = {ZERO8, ZERO8, ZERO8};
            gemm_block<3>(acc, h2 + 8192,  256, wp + 647168, 8,  ct, lane);
            gemm_block<3>(acc, xs + 4096,  128, wp + 753664, 8,  ct, lane);
            out_store<3>(acc, y, row0, 256, ct * 16, stats, 512, lane);
        } else {
            const int ct = t - 16;
            v8f acc[5] = {ZERO8, ZERO8, ZERO8, ZERO8, ZERO8};
            gemm_block<5>(acc, h2 + 20480, 128, wp + 679936, 4,  ct, lane);
            gemm_block<5>(acc, xs + 10240, 64,  wp + 770048, 4,  ct, lane);
            out_store<5>(acc, y, row0, 640, ct * 16, stats, 640, lane);
        }
    }
}

__global__ void zero_stats(float* __restrict__ stats)
{
    for (int i = threadIdx.x; i < 704; i += blockDim.x) stats[i] = 0.f;
}

// BatchNorm finalize, in place on d_out.
__global__ void bn_finalize(float* __restrict__ y,
                            const float* __restrict__ stats,
                            const float* __restrict__ bn_w,
                            const float* __restrict__ bn_b)
{
    const int idx = blockIdx.x * blockDim.x + threadIdx.x;   // < 32768*960
    const int col = idx % 960;
    float v = y[idx];
    float o;
    if (col < 256) {
        const float mu  = stats[col] * (1.f / 32768.f);
        const float var = stats[256 + col] * (1.f / 32768.f) - mu * mu;
        o = (v - mu) * (bn_w[col] * rsqrtf(var + 1e-5f)) + bn_b[col];
    } else if (col < 640) {
        const int ch = (col - 256) / 3;
        const float n2 = stats[512 + ch] * (1.f / (3.f * 32768.f));
        o = v * bn_w[256 + ch] * rsqrtf(n2 + 1e-5f);
    } else {
        const int ch = (col - 640) / 5;
        const float n2 = stats[640 + ch] * (1.f / (5.f * 32768.f));
        o = v * bn_w[384 + ch] * rsqrtf(n2 + 1e-5f);
    }
    y[idx] = o;
}

extern "C" void kernel_launch(void* const* d_in, const int* in_sizes, int n_in,
                              void* d_out, int out_size, void* d_ws, size_t ws_size,
                              hipStream_t stream)
{
    const float* x   = (const float*)d_in[0];
    const float* w1  = (const float*)d_in[1];
    const float* w2  = (const float*)d_in[2];
    const float* w3  = (const float*)d_in[3];
    const float* wb  = (const float*)d_in[4];
    const float* bnw = (const float*)d_in[5];
    const float* bnb = (const float*)d_in[6];
    float*  y     = (float*)d_out;
    __bf16* wpk   = (__bf16*)d_ws;                         // 774144 bf16
    float*  stats = (float*)((char*)d_ws + 1548288);       // 704 f32

    pack_weights<<<dim3(1024, 12, 1), 256, 0, stream>>>(w1, w2, w3, wb, wpk);
    zero_stats<<<1, 256, 0, stream>>>(stats);
    ff_fused<<<2048, 256, 153600, stream>>>(x, wpk, y, stats);
    bn_finalize<<<122880, 256, 0, stream>>>(y, stats, bnw, bnb);
}
